// seq2seq_pred_32263794328386
// MI455X (gfx1250) — matrix operand order repaired and checked by reading, not compiled, not run on hardware
//
#include <hip/hip_runtime.h>

#define BB 256
#define SS 512
#define RR 128
#define HH 512
#define NBLK 4   // N-tiles per wave (16x64 of C per wave)

typedef __attribute__((ext_vector_type(16))) _Float16 v16h;
typedef __attribute__((ext_vector_type(8)))  _Float16 v8h;
typedef __attribute__((ext_vector_type(8)))  float    v8f;

union V16H { v16h v; v8h h2[2]; };
union V8F  { v8f v; float f[8]; };

// ---------------------------------------------------------------------------
// WMMA GEMM: C[M,N] = A[M,K] * B[N,K]^T  (+ bias[n]) (+ x[m]*u[n] rank-1 term)
// Register-blocked: one wave32 computes a 16x(16*NBLK) strip of C.
// Per K-step (32): load A fragment once, NBLK B fragments, NBLK back-to-back
// v_wmma_f32_16x16x32_f16 into independent accumulators (ILP for XDL pipe).
// Fragment layouts as loaded here (wave32, same contiguous K order for A and B):
//   A 16x32 f16 : lanes 0-15 row M=lane, halves = K[0..8)+K[8..16);
//                 lanes 16-31 row M=lane-16, halves = K[16..24)+K[24..32)
//   B 32x16 f16 : lane n holds column n; lanes 0-15 K[0..16), lanes16-31 K[16..32)
//   C 16x16 f32 : VGPR r -> row (lane>=16 ? 8+r : r), col = lane&15
// Requires N % (16*NBLK) == 0 (true here: N in {512, 2048}).
// ---------------------------------------------------------------------------
__global__ void wmma_gemm_bt(const _Float16* __restrict__ A, int lda,
                             const _Float16* __restrict__ Bm, int ldb,
                             void* __restrict__ Cout, int ldc, int out_f16,
                             int M, int N, int K,
                             const float* __restrict__ bias,
                             const float* __restrict__ xvec, int xstride,
                             const float* __restrict__ uvec)
{
    const int lane = threadIdx.x & 31;
    const int wave = (blockIdx.x * blockDim.x + threadIdx.x) >> 5;
    const int ntn   = N / (16 * NBLK);           // N strips
    const int tiles = (M >> 4) * ntn;
    if (wave >= tiles) return;                   // wave-uniform: EXEC all-1 past here
    const int tm  = wave / ntn;
    const int tn0 = (wave - tm * ntn) * NBLK;    // first N-tile of this strip

    const int row_a = tm * 16 + (lane & 15);
    const int row_b = tn0 * 16 + (lane & 15);
    const int ka = (lane >> 4) * 16;             // A half-lane K split
    const int kb = (lane >> 4) * 16;             // B half-lane K split

    const _Float16* ap = A  + (size_t)row_a * lda + ka;
    const _Float16* bp = Bm + (size_t)row_b * ldb + kb;
    const size_t bstep = (size_t)16 * ldb;       // next N-tile of B

    v8f acc0 = {}, acc1 = {}, acc2 = {}, acc3 = {};
#pragma unroll 2
    for (int k0 = 0; k0 < K; k0 += 32) {
        V16H a;
        a.h2[0] = *reinterpret_cast<const v8h*>(ap + k0);        // K = k0+ka .. +8
        a.h2[1] = *reinterpret_cast<const v8h*>(ap + k0 + 8);    // K = k0+8+ka .. +8
        V16H b0, b1, b2, b3;
        b0.v = *reinterpret_cast<const v16h*>(bp + k0);
        b1.v = *reinterpret_cast<const v16h*>(bp + bstep + k0);
        b2.v = *reinterpret_cast<const v16h*>(bp + 2 * bstep + k0);
        b3.v = *reinterpret_cast<const v16h*>(bp + 3 * bstep + k0);
        acc0 = __builtin_amdgcn_wmma_f32_16x16x32_f16(false, a.v, false, b0.v,
                                                      (short)0, acc0, false, false);
        acc1 = __builtin_amdgcn_wmma_f32_16x16x32_f16(false, a.v, false, b1.v,
                                                      (short)0, acc1, false, false);
        acc2 = __builtin_amdgcn_wmma_f32_16x16x32_f16(false, a.v, false, b2.v,
                                                      (short)0, acc2, false, false);
        acc3 = __builtin_amdgcn_wmma_f32_16x16x32_f16(false, a.v, false, b3.v,
                                                      (short)0, acc3, false, false);
    }

    const int crow0 = tm * 16 + (lane >> 4) * 8;
    float xr[8];
#pragma unroll
    for (int r = 0; r < 8; ++r)
        xr[r] = xvec ? xvec[(size_t)(crow0 + r) * xstride] : 0.0f;

    v8f accs[NBLK] = { acc0, acc1, acc2, acc3 };
#pragma unroll
    for (int j = 0; j < NBLK; ++j) {
        const int ccol = (tn0 + j) * 16 + (lane & 15);
        const float bn = bias ? bias[ccol] : 0.0f;
        const float un = uvec ? uvec[ccol] : 0.0f;
        V8F res; res.v = accs[j];
#pragma unroll
        for (int r = 0; r < 8; ++r) {
            const int crow = crow0 + r;
            float val = res.f[r] + bn + xr[r] * un;
            if (out_f16) ((_Float16*)Cout)[(size_t)crow * ldc + ccol] = (_Float16)val;
            else         ((float*)   Cout)[(size_t)crow * ldc + ccol] = val;
        }
    }
}

// ---------------------------------------------------------------------------
// Pointwise LSTM cell: gates[B,4H] (i,f,g,o) + c -> c', h' (f16 out, f32 opt)
// ---------------------------------------------------------------------------
__global__ void lstm_pointwise(const float* __restrict__ gates,
                               float* __restrict__ c,
                               float* __restrict__ h32,
                               _Float16* __restrict__ h16, int hstride,
                               int H_, int total)
{
    int idx = blockIdx.x * blockDim.x + threadIdx.x;
    if (idx >= total) return;
    int b  = idx / H_;
    int hh = idx - b * H_;
    const float* g = gates + (size_t)b * 4 * H_;
    float ig = 1.0f / (1.0f + __expf(-g[hh]));
    float fg = 1.0f / (1.0f + __expf(-g[H_ + hh]));
    float gg = tanhf(g[2 * H_ + hh]);
    float og = 1.0f / (1.0f + __expf(-g[3 * H_ + hh]));
    float cn = fg * c[idx] + ig * gg;
    c[idx] = cn;
    float hn = og * tanhf(cn);
    if (h32) h32[idx] = hn;
    h16[(size_t)b * hstride + hh] = (_Float16)hn;
}

// ---------------------------------------------------------------------------
// Fused attention: scores = q . kproj, softmax over S, context = w . hxs
// One 256-thread block per batch element. Writes context into decA[:, H:2H).
// ---------------------------------------------------------------------------
__global__ void attn_kernel(const float* __restrict__ q,
                            const _Float16* __restrict__ kproj,
                            const _Float16* __restrict__ hxs,
                            _Float16* __restrict__ decA,
                            int S_, int H_)
{
    __shared__ float sh_q[HH];
    __shared__ float sh_w[SS];
    __shared__ float red[256];
    const int b = blockIdx.x;
    const int t = threadIdx.x;

    for (int i = t; i < H_; i += 256) sh_q[i] = q[(size_t)b * H_ + i];
    __syncthreads();

    const _Float16* kb = kproj + (size_t)b * S_ * H_;
    for (int s = t; s < S_; s += 256) {
        const _Float16* kr = kb + (size_t)s * H_;
        float acc = 0.0f;
        for (int i = 0; i < H_; i += 8) {
            v8h kk = *reinterpret_cast<const v8h*>(kr + i);
#pragma unroll
            for (int j = 0; j < 8; ++j) acc += (float)kk[j] * sh_q[i + j];
        }
        sh_w[s] = acc;
    }
    __syncthreads();

    // softmax (max)
    float m = -1e30f;
    for (int s = t; s < S_; s += 256) m = fmaxf(m, sh_w[s]);
    red[t] = m; __syncthreads();
    for (int off = 128; off; off >>= 1) {
        if (t < off) red[t] = fmaxf(red[t], red[t + off]);
        __syncthreads();
    }
    const float mx = red[0];
    __syncthreads();
    // softmax (exp + sum)
    float sum = 0.0f;
    for (int s = t; s < S_; s += 256) { float e = __expf(sh_w[s] - mx); sh_w[s] = e; sum += e; }
    red[t] = sum; __syncthreads();
    for (int off = 128; off; off >>= 1) {
        if (t < off) red[t] += red[t + off];
        __syncthreads();
    }
    const float inv = 1.0f / red[0];
    __syncthreads();

    // context (adjacent threads read adjacent h -> coalesced)
    const _Float16* hb = hxs + (size_t)b * S_ * H_;
    for (int hh = t; hh < H_; hh += 256) {
        float acc = 0.0f;
        for (int s = 0; s < S_; ++s) acc += sh_w[s] * (float)hb[(size_t)s * H_ + hh];
        decA[(size_t)b * 2 * H_ + H_ + hh] = (_Float16)(acc * inv);
    }
}

// ---------------------------------------------------------------------------
// Output projection: out[b,t] = dot(h32[b,:], out_W) + out_b  (1 wave per b)
// ---------------------------------------------------------------------------
__global__ void outproj_kernel(const float* __restrict__ h32,
                               const float* __restrict__ outW,
                               const float* __restrict__ outb,
                               float* __restrict__ out,
                               int H_, int T_, int tstep)
{
    const int lane = threadIdx.x & 31;
    const int b = blockIdx.x * (blockDim.x >> 5) + (threadIdx.x >> 5);
    float acc = 0.0f;
    for (int i = lane; i < H_; i += 32) acc += h32[(size_t)b * H_ + i] * outW[i];
#pragma unroll
    for (int off = 16; off; off >>= 1) acc += __shfl_xor(acc, off, 32);
    if (lane == 0) out[(size_t)b * T_ + tstep] = acc + outb[0];
}

// ------------------------------- setup kernels ------------------------------
__global__ void cvt_f16(const float* __restrict__ src, _Float16* __restrict__ dst, int n)
{
    int i = blockIdx.x * blockDim.x + threadIdx.x;
    if (i < n) dst[i] = (_Float16)src[i];
}

__global__ void zero_state(_Float16* __restrict__ h0, float* __restrict__ c, int n)
{
    int i = blockIdx.x * blockDim.x + threadIdx.x;
    if (i < n) { h0[i] = (_Float16)0.0f; c[i] = 0.0f; }
}

// u = enc_Wih @ embed_W ; v = enc_Wih @ embed_b + enc_bih + enc_bhh  (rank-1 fold)
__global__ void build_uv(const float* __restrict__ eWih, const float* __restrict__ embW,
                         const float* __restrict__ embb, const float* __restrict__ bih,
                         const float* __restrict__ bhh,
                         float* __restrict__ u, float* __restrict__ v, int G, int R_)
{
    int g = blockIdx.x * blockDim.x + threadIdx.x;
    if (g >= G) return;
    float su = 0.0f, sv = 0.0f;
    for (int r = 0; r < R_; ++r) {
        float w = eWih[(size_t)g * R_ + r];
        su += w * embW[r];
        sv += w * embb[r];
    }
    u[g] = su;
    v[g] = sv + bih[g] + bhh[g];
}

// Wcomb[g, 0:H) = dec_Wih[g, 0:H) + dec_Whh[g,:] ; Wcomb[g, H:2H) = dec_Wih[g, H:2H)
__global__ void build_wcomb(const float* __restrict__ dWih, const float* __restrict__ dWhh,
                            _Float16* __restrict__ wc,
                            const float* __restrict__ dbih, const float* __restrict__ dbhh,
                            float* __restrict__ dbias, int G, int H_)
{
    int i = blockIdx.x * blockDim.x + threadIdx.x;
    int n = G * 2 * H_;
    if (i < n) {
        int g = i / (2 * H_);
        int k = i - g * 2 * H_;
        float val = dWih[i] + (k < H_ ? dWhh[(size_t)g * H_ + k] : 0.0f);
        wc[i] = (_Float16)val;
    }
    if (i < G) dbias[i] = dbih[i] + dbhh[i];
}

__global__ void copy_h_init(const _Float16* __restrict__ hxs_last,
                            _Float16* __restrict__ decA, int H_, int n)
{
    int i = blockIdx.x * blockDim.x + threadIdx.x;
    if (i >= n) return;
    int b = i / H_, hh = i - b * H_;
    decA[(size_t)b * 2 * H_ + hh] = hxs_last[(size_t)b * SS * H_ + hh];
}

// ---------------------------------------------------------------------------
extern "C" void kernel_launch(void* const* d_in, const int* in_sizes, int n_in,
                              void* d_out, int out_size, void* d_ws, size_t ws_size,
                              hipStream_t stream)
{
    (void)in_sizes; (void)n_in; (void)ws_size;

    const float* enc_inputs = (const float*)d_in[0];
    // d_in[1] = out_seq_len scalar; T derived from out_size instead (host-known)
    const float* embed_W = (const float*)d_in[2];
    const float* embed_b = (const float*)d_in[3];
    const float* enc_Wih = (const float*)d_in[4];
    const float* enc_Whh = (const float*)d_in[5];
    const float* enc_bih = (const float*)d_in[6];
    const float* enc_bhh = (const float*)d_in[7];
    const float* dec_Wih = (const float*)d_in[8];
    const float* dec_Whh = (const float*)d_in[9];
    const float* dec_bih = (const float*)d_in[10];
    const float* dec_bhh = (const float*)d_in[11];
    const float* Wq      = (const float*)d_in[12];
    const float* bq      = (const float*)d_in[13];
    const float* Wk      = (const float*)d_in[14];
    const float* bk      = (const float*)d_in[15];
    const float* out_W   = (const float*)d_in[16];
    const float* out_b   = (const float*)d_in[17];
    float* out = (float*)d_out;
    const int T = out_size / BB;   // 96

    // ---- workspace carve-up (256B aligned) ----
    char* w = (char*)d_ws;
    size_t off = 0;
    auto alloc = [&](size_t bytes) -> char* {
        char* p = w + off;
        off += (bytes + 255) & ~(size_t)255;
        return p;
    };
    _Float16* hxs   = (_Float16*)alloc((size_t)BB * SS * HH * 2);   // encoder h history (f16)
    _Float16* kproj = (_Float16*)alloc((size_t)BB * SS * HH * 2);   // key projection (f16)
    float*    gates = (float*)   alloc((size_t)BB * 4 * HH * 4);
    float*    cbuf  = (float*)   alloc((size_t)BB * HH * 4);
    float*    h32   = (float*)   alloc((size_t)BB * HH * 4);
    float*    qbuf  = (float*)   alloc((size_t)BB * HH * 4);
    _Float16* decA  = (_Float16*)alloc((size_t)BB * 2 * HH * 2);    // [h | context] f16
    _Float16* h0    = (_Float16*)alloc((size_t)BB * HH * 2);
    _Float16* whh16 = (_Float16*)alloc((size_t)4 * HH * HH * 2);
    _Float16* wq16  = (_Float16*)alloc((size_t)HH * HH * 2);
    _Float16* wk16  = (_Float16*)alloc((size_t)HH * HH * 2);
    _Float16* wcomb = (_Float16*)alloc((size_t)4 * HH * 2 * HH * 2);
    float*    u     = (float*)   alloc(4 * HH * 4);
    float*    vv    = (float*)   alloc(4 * HH * 4);
    float*    dbias = (float*)   alloc(4 * HH * 4);

    auto gemm = [&](const _Float16* A, int lda, const _Float16* Bm, int ldb,
                    void* C, int ldc, int outf16, int M, int N, int K,
                    const float* bias, const float* xv, int xs, const float* uv) {
        int tiles  = (M / 16) * (N / (16 * NBLK));
        int blocks = (tiles * 32 + 255) / 256;
        wmma_gemm_bt<<<blocks, 256, 0, stream>>>(A, lda, Bm, ldb, C, ldc, outf16,
                                                 M, N, K, bias, xv, xs, uv);
    };

    // ---- one-time prep (cheap; weights end up L2-resident in f16) ----
    cvt_f16<<<(4 * HH * HH + 255) / 256, 256, 0, stream>>>(enc_Whh, whh16, 4 * HH * HH);
    cvt_f16<<<(HH * HH + 255) / 256, 256, 0, stream>>>(Wq, wq16, HH * HH);
    cvt_f16<<<(HH * HH + 255) / 256, 256, 0, stream>>>(Wk, wk16, HH * HH);
    build_wcomb<<<(4 * HH * 2 * HH + 255) / 256, 256, 0, stream>>>(
        dec_Wih, dec_Whh, wcomb, dec_bih, dec_bhh, dbias, 4 * HH, HH);
    build_uv<<<(4 * HH + 255) / 256, 256, 0, stream>>>(
        enc_Wih, embed_W, embed_b, enc_bih, enc_bhh, u, vv, 4 * HH, RR);
    zero_state<<<(BB * HH + 255) / 256, 256, 0, stream>>>(h0, cbuf, BB * HH);

    // ---- encoder: 512 sequential steps; gates = h@Whh^T + e[b,s]*u + v ----
    for (int s = 0; s < SS; ++s) {
        const _Float16* hA = (s == 0) ? h0 : (hxs + (size_t)(s - 1) * HH);
        int lda = (s == 0) ? HH : SS * HH;   // hxs[:,s,:] view has row stride S*H
        gemm(hA, lda, whh16, HH, gates, 4 * HH, 0,
             BB, 4 * HH, HH, vv, enc_inputs + s, SS, u);
        lstm_pointwise<<<(BB * HH + 255) / 256, 256, 0, stream>>>(
            gates, cbuf, (float*)nullptr, hxs + (size_t)s * HH, SS * HH, HH, BB * HH);
    }

    // ---- key projection: one big GEMM  kproj = hxs @ Wk^T + bk  (f16 out) ----
    gemm(hxs, HH, wk16, HH, kproj, HH, 1, BB * SS, HH, HH, bk, nullptr, 0, nullptr);

    // ---- decoder init: h = last encoder h ----
    copy_h_init<<<(BB * HH + 255) / 256, 256, 0, stream>>>(
        hxs + (size_t)(SS - 1) * HH, decA, HH, BB * HH);

    // ---- decoder: T sequential steps ----
    for (int t = 0; t < T; ++t) {
        // q = h @ Wq^T + bq  (reads decA[:, 0:H))
        gemm(decA, 2 * HH, wq16, HH, qbuf, HH, 0, BB, HH, HH, bq, nullptr, 0, nullptr);
        // scores/softmax/context fused; context -> decA[:, H:2H)
        attn_kernel<<<BB, 256, 0, stream>>>(qbuf, kproj, hxs, decA, SS, HH);
        // gates = [h,ctx] @ Wcomb^T + (bih+bhh)   (K = 2H, folds Wih+Whh)
        gemm(decA, 2 * HH, wcomb, 2 * HH, gates, 4 * HH, 0,
             BB, 4 * HH, 2 * HH, dbias, nullptr, 0, nullptr);
        lstm_pointwise<<<(BB * HH + 255) / 256, 256, 0, stream>>>(
            gates, cbuf, h32, decA, 2 * HH, HH, BB * HH);
        outproj_kernel<<<BB / 8, 256, 0, stream>>>(h32, out_W, out_b, out, HH, T, t);
    }
}
